// DTGNN_1795296330250
// MI455X (gfx1250) — compile-verified, matrix operand order checked by reading, not run on hardware
//
#include <hip/hip_runtime.h>
#include <hip/hip_bf16.h>
#include <math.h>

typedef __attribute__((ext_vector_type(16))) _Float16 v16h;
typedef __attribute__((ext_vector_type(8)))  _Float16 v8h;
typedef __attribute__((ext_vector_type(4)))  _Float16 v4h;
typedef __attribute__((ext_vector_type(8)))  float    v8f;
typedef __attribute__((ext_vector_type(4)))  float    v4f;

#define T_STEPS 12
#define N_NODES 30000
#define M_ROWS  30000
#define E_EDGES 960000

// ---------------------------------------------------------------------------
// WMMA helpers (CDNA5 16x16x32 f16 -> f32)
// ---------------------------------------------------------------------------
__device__ __forceinline__ v8f wmma_f16(v16h a, v16h b, v8f c) {
  // 8 args: (neg_a, A, neg_b, B, c_mod, C, reuse_a, reuse_b)
  return __builtin_amdgcn_wmma_f32_16x16x32_f16(false, a, false, b, (short)0, c,
                                                false, false);
}

__device__ __forceinline__ v8f zero8() {
  v8f z;
#pragma unroll
  for (int e = 0; e < 8; ++e) z[e] = 0.f;
  return z;
}

// A-fragment (16x32, f16) from LDS row-major [row][k] matrix.
// ISA layout: lane holds row=lane&15; K runs {base..base+7, base+16..base+23},
// base = (lane>>4)*8.  Two 16B ds loads per fragment.
__device__ __forceinline__ v16h load_a_lds(const _Float16* sm, int stride,
                                           int kbase, int lane) {
  int row = lane & 15;
  const _Float16* p = sm + row * stride + kbase + ((lane >> 4) << 3);
  v8h r0 = *(const v8h*)(p);
  v8h r1 = *(const v8h*)(p + 16);
  v16h a;
#pragma unroll
  for (int j = 0; j < 8; ++j) { a[j] = r0[j]; a[j + 8] = r1[j]; }
  return a;
}

// K index inside a 32-wide chunk for (lane, element j) -- shared by packers.
__device__ __forceinline__ int kmap(int lane, int j) {
  return ((lane >> 4) << 3) + (j < 8 ? j : j + 8);
}

// ---------------------------------------------------------------------------
// Weight pre-swizzle: W (dirs=2, 192, Cin) fp32 -> B-fragments f16.
// Fragment f = kc*12 + nt;  element idx = ((d*KC*12 + f)*32 + lane)*16 + j.
// B[k][n] = W[n][k];  n = nt*16 + (lane&15);  k = kc*32 + kmap(lane,j).
// ---------------------------------------------------------------------------
__global__ void k_pack_w_dtg(const float* __restrict__ W, _Float16* __restrict__ dst,
                             int Cin, int KC) {
  int idx = blockIdx.x * blockDim.x + threadIdx.x;
  int total = 2 * KC * 12 * 512;
  if (idx >= total) return;
  int j = idx & 15;
  int lane = (idx >> 4) & 31;
  int f = (idx >> 9) % (KC * 12);
  int d = idx / (KC * 12 * 512);
  int kc = f / 12, nt = f % 12;
  int n = nt * 16 + (lane & 15);
  int k = kc * 32 + kmap(lane, j);
  float v = (k < Cin) ? W[((size_t)d * 192 + n) * Cin + k] : 0.f;
  dst[idx] = (_Float16)v;
}

// G[t,i][j][o] = sum_k cheb_W[i,j,k] * conv_W[o,t,0,k], packed in B-fragment
// order.  72 fragments (kc = t*6 + r; r: i=r/2, j0=(r%2)*32), cols padded to 16.
__global__ void k_pack_G_dtg(const float* __restrict__ chebW,
                             const float* __restrict__ convW,
                             _Float16* __restrict__ dst) {
  int idx = blockIdx.x * blockDim.x + threadIdx.x;
  if (idx >= 72 * 512) return;
  int j = idx & 15;
  int lane = (idx >> 4) & 31;
  int kc = idx >> 9;
  int o = lane & 15;
  int kk = kc * 32 + kmap(lane, j);
  int t = kk / 192, rr = kk % 192, i = rr / 64, jj = rr % 64;
  float g = 0.f;
  if (o < 12) {
    const float* wi = chebW + ((size_t)i * 64 + jj) * 128;
    const float* cw = convW + ((size_t)o * 12 + t) * 128;
#pragma unroll 8
    for (int k = 0; k < 128; ++k) g += wi[k] * cw[k];
  }
  dst[idx] = (_Float16)g;
}

__global__ void k_biasO_dtg(const float* __restrict__ chebB,
                            const float* __restrict__ convW,
                            const float* __restrict__ convB,
                            float* __restrict__ biasO) {
  int o = threadIdx.x;
  if (o >= 12) return;
  float s = convB[o];
  for (int t = 0; t < 12; ++t) {
    const float* cw = convW + ((size_t)o * 12 + t) * 128;
    for (int k = 0; k < 128; ++k) s += chebB[k] * cw[k];
  }
  biasO[o] = s;
}

// ---------------------------------------------------------------------------
// BiGRU layer, KC = number of 32-wide K chunks of the input (compile-time so
// all fragment arrays are statically register-allocated; no v_movrels).
// grid = (ceil(M/64), ndirs), block = 128 (4 independent waves, 16 rows each).
// h kept in WMMA C-layout accumulators; re-staged through LDS as f16 each step
// to be consumed as A-fragments.  Weights read as pre-swizzled B-fragments
// straight from global (L2-resident).
// ---------------------------------------------------------------------------
template <int KC>
__global__ void __launch_bounds__(128) k_bigru_dtg(
    const float* __restrict__ in, long m_stride, long t_stride,
    const _Float16* __restrict__ packIh, const _Float16* __restrict__ packHh,
    const float* __restrict__ bih, const float* __restrict__ bhh,
    float* __restrict__ out, int out_stride, int out_dirmul, int dir0) {
  constexpr int Cin = (KC == 1) ? 16 : KC * 32;  // layer0: 16 of 32; layer1: 128
  constexpr int Cpad = KC * 32;
  __shared__ __align__(16) _Float16 smem[4 * (16 * 128 + 16 * 64)];

  int wave = threadIdx.x >> 5;
  int lane = threadIdx.x & 31;
  int tile = blockIdx.x * 4 + wave;
  if (tile * 16 >= M_ROWS) return;  // uniform per-wave exit: EXEC stays all-1
  int mbase = tile * 16;
  int d = dir0 + blockIdx.y;
  int half = lane >> 4;
  int colL = lane & 15;

  _Float16* smx = smem + wave * (16 * 128 + 16 * 64);
  _Float16* smh = smx + 16 * 128;
  const _Float16* pIh = packIh + (size_t)d * KC * 12 * 512;
  const _Float16* pHh = packHh + (size_t)d * 2 * 12 * 512;
  const float* bi = bih + d * 192;
  const float* bh = bhh + d * 192;

  __builtin_prefetch(pIh + lane * 16, 0, 1);
  __builtin_prefetch(pHh + lane * 16, 0, 1);

  // zero K-padding of the x staging area (x stage only writes c < Cin)
  if (Cin < Cpad) {
    for (int idx = lane; idx < 16 * Cpad; idx += 32)
      if ((idx % Cpad) >= Cin) smx[idx] = (_Float16)0.f;
  }

  v8f hT[4];
#pragma unroll
  for (int g = 0; g < 4; ++g) hT[g] = zero8();

  for (int s = 0; s < T_STEPS; ++s) {
    int t = (d == 0) ? s : (T_STEPS - 1 - s);

    // stage x_t tile (16 x Cin) -> LDS f16, vectorized float4 -> b64 stores
    {
      constexpr int C4 = Cin / 4;
#pragma unroll
      for (int idx = lane; idx < 16 * C4; idx += 32) {
        int r = idx / C4, c4 = idx % C4;
        int m = mbase + r;
        if (m >= M_ROWS) m = M_ROWS - 1;
        v4f f = *(const v4f*)(in + (size_t)m * m_stride + (size_t)t * t_stride +
                              c4 * 4);
        v4h h4;
#pragma unroll
        for (int q = 0; q < 4; ++q) h4[q] = (_Float16)f[q];
        *(v4h*)(smx + r * Cpad + c4 * 4) = h4;
      }
    }
    // stage h (C layout -> row-major f16 [16][64])
#pragma unroll
    for (int g = 0; g < 4; ++g)
#pragma unroll
      for (int e = 0; e < 8; ++e)
        smh[(e + half * 8) * 64 + g * 16 + colL] = (_Float16)hT[g][e];

    // A fragments (per-wave LDS region; same-wave DS ops are in order)
    v16h ax[KC];
#pragma unroll
    for (int kc = 0; kc < KC; ++kc) ax[kc] = load_a_lds(smx, Cpad, kc * 32, lane);
    v16h ah0 = load_a_lds(smh, 64, 0, lane);
    v16h ah1 = load_a_lds(smh, 64, 32, lane);

#pragma unroll
    for (int g = 0; g < 4; ++g) {
      v8f aR = zero8(), aZ = zero8(), aI = zero8(), aH = zero8();
      int ntR = g, ntZ = 4 + g, ntN = 8 + g;
#pragma unroll
      for (int kc = 0; kc < KC; ++kc) {
        const _Float16* base = pIh + (size_t)(kc * 12) * 512 + (size_t)lane * 16;
        aR = wmma_f16(ax[kc], *(const v16h*)(base + (size_t)ntR * 512), aR);
        aZ = wmma_f16(ax[kc], *(const v16h*)(base + (size_t)ntZ * 512), aZ);
        aI = wmma_f16(ax[kc], *(const v16h*)(base + (size_t)ntN * 512), aI);
      }
#pragma unroll
      for (int kc = 0; kc < 2; ++kc) {
        const _Float16* base = pHh + (size_t)(kc * 12) * 512 + (size_t)lane * 16;
        v16h ah = kc ? ah1 : ah0;
        aR = wmma_f16(ah, *(const v16h*)(base + (size_t)ntR * 512), aR);
        aZ = wmma_f16(ah, *(const v16h*)(base + (size_t)ntZ * 512), aZ);
        aH = wmma_f16(ah, *(const v16h*)(base + (size_t)ntN * 512), aH);
      }
      int c0 = g * 16 + colL;
      float br = bi[c0] + bh[c0];
      float bz = bi[64 + c0] + bh[64 + c0];
      float bin = bi[128 + c0];
      float bhn = bh[128 + c0];
#pragma unroll
      for (int e = 0; e < 8; ++e) {
        float r = 1.f / (1.f + __expf(-(aR[e] + br)));
        float z = 1.f / (1.f + __expf(-(aZ[e] + bz)));
        float nn = tanhf(aI[e] + bin + r * (aH[e] + bhn));
        hT[g][e] = (1.f - z) * nn + z * hT[g][e];
      }
#pragma unroll
      for (int e = 0; e < 8; ++e) {
        int m = mbase + e + half * 8;
        if (m < M_ROWS)
          out[((size_t)m * T_STEPS + t) * out_stride + out_dirmul * d + c0] =
              hT[g][e];
      }
    }
  }
}

// ---------------------------------------------------------------------------
// LayerNorm of backward half -> time-major xs[t][m][64].  One wave per row.
// ---------------------------------------------------------------------------
__global__ void __launch_bounds__(128) k_ln_dtg(const float* __restrict__ out1b,
                                                const float* __restrict__ scale,
                                                const float* __restrict__ bias,
                                                float* __restrict__ xs) {
  long wg = (long)blockIdx.x * 4 + (threadIdx.x >> 5);
  int lane = threadIdx.x & 31;
  if (wg >= (long)M_ROWS * T_STEPS) return;
  int m = (int)(wg / T_STEPS), t = (int)(wg % T_STEPS);
  const float* src = out1b + ((size_t)m * T_STEPS + t) * 64;
  float x0 = src[lane], x1 = src[lane + 32];
  float s = x0 + x1, sq = x0 * x0 + x1 * x1;
#pragma unroll
  for (int o = 16; o; o >>= 1) {
    s += __shfl_xor(s, o, 32);
    sq += __shfl_xor(sq, o, 32);
  }
  float mu = s * (1.f / 64.f);
  float var = sq * (1.f / 64.f) - mu * mu;
  float inv = rsqrtf(var + 1e-5f);
  float* dst = xs + ((size_t)t * M_ROWS + m) * 64;
  dst[lane] = (x0 - mu) * inv * scale[lane] + bias[lane];
  dst[lane + 32] = (x1 - mu) * inv * scale[lane + 32] + bias[lane + 32];
}

// ---------------------------------------------------------------------------
// Graph utility kernels (batched over t where possible)
// ---------------------------------------------------------------------------
__global__ void k_zero_dtg(float* p, long n) {
  long i = (long)blockIdx.x * blockDim.x + threadIdx.x;
  if (i < n) p[i] = 0.f;
}
__global__ void k_fill_dtg(float* p, long n, float v) {
  long i = (long)blockIdx.x * blockDim.x + threadIdx.x;
  if (i < n) p[i] = v;
}
__global__ void k_deg_dtg(const int* __restrict__ ei, float* __restrict__ deg) {
  long i = (long)blockIdx.x * blockDim.x + threadIdx.x;
  if (i >= (long)T_STEPS * E_EDGES) return;
  int t = (int)(i / E_EDGES);
  long e = i % E_EDGES;
  int row = ei[((long)t * 2 + 0) * E_EDGES + e];
  atomicAdd(&deg[(long)t * M_ROWS + row], 1.f);
}
__global__ void k_dinv_dtg(const float* __restrict__ deg, float* __restrict__ dinv) {
  long i = (long)blockIdx.x * blockDim.x + threadIdx.x;
  if (i >= (long)T_STEPS * M_ROWS) return;
  float d = deg[i];
  dinv[i] = d > 0.f ? rsqrtf(fmaxf(d, 1.f)) : 0.f;
}
__global__ void k_pa_dtg(const float* __restrict__ deg, const float* __restrict__ v,
                         float* __restrict__ w) {
  long i = (long)blockIdx.x * blockDim.x + threadIdx.x;
  if (i < (long)T_STEPS * M_ROWS) w[i] = deg[i] * v[i];
}
__global__ void k_pb_dtg(const int* __restrict__ ei, const float* __restrict__ v,
                         float* __restrict__ w) {
  long i = (long)blockIdx.x * blockDim.x + threadIdx.x;
  if (i >= (long)T_STEPS * E_EDGES) return;
  int t = (int)(i / E_EDGES);
  long e = i % E_EDGES;
  int row = ei[((long)t * 2 + 0) * E_EDGES + e];
  int col = ei[((long)t * 2 + 1) * E_EDGES + e];
  atomicAdd(&w[(long)t * M_ROWS + row], -v[(long)t * M_ROWS + col]);
}
// dst[t] += sum_i a[t*M+i]*b[t*M+i]   (sumsq when a==b, dot otherwise)
__global__ void k_red_dtg(const float* __restrict__ a, const float* __restrict__ b,
                          float* __restrict__ dst) {
  int t = blockIdx.y;
  __shared__ float red[256];
  float s = 0.f;
  for (long i = (long)blockIdx.x * 256 + threadIdx.x; i < M_ROWS;
       i += (long)gridDim.x * 256)
    s += a[(long)t * M_ROWS + i] * b[(long)t * M_ROWS + i];
  red[threadIdx.x] = s;
  __syncthreads();
  for (int o = 128; o; o >>= 1) {
    if (threadIdx.x < o) red[threadIdx.x] += red[threadIdx.x + o];
    __syncthreads();
  }
  if (threadIdx.x == 0) atomicAdd(&dst[t], red[0]);
}
__global__ void k_scale_dtg(const float* __restrict__ w, const float* __restrict__ sums,
                            float* __restrict__ v) {
  long i = (long)blockIdx.x * blockDim.x + threadIdx.x;
  if (i >= (long)T_STEPS * M_ROWS) return;
  int t = (int)(i / M_ROWS);
  v[i] = w[i] * (1.f / (sqrtf(sums[t]) + 1e-12f));
}

// ---------------------------------------------------------------------------
// Chebyshev per-t kernels
// ---------------------------------------------------------------------------
__global__ void k_cheb_init1_dtg(const float* __restrict__ xs_t,
                                 const float* __restrict__ lam,
                                 float* __restrict__ Tx1, int t) {
  long i = (long)blockIdx.x * blockDim.x + threadIdx.x;
  if (i >= (long)M_ROWS * 64) return;
  float diag = 2.f / lam[t] - 1.f;
  Tx1[i] = diag * xs_t[i];
}
__global__ void k_cheb_init2_dtg(const float* __restrict__ xs_t,
                                 const float* __restrict__ Tx1,
                                 const float* __restrict__ lam,
                                 float* __restrict__ Tx2, int t) {
  long i = (long)blockIdx.x * blockDim.x + threadIdx.x;
  if (i >= (long)M_ROWS * 64) return;
  float diag = 2.f / lam[t] - 1.f;
  Tx2[i] = 2.f * diag * Tx1[i] - xs_t[i];
}
// dst[row,:] += mul * (-2/lam) * dinv[row]*dinv[col] * src[col,:]; wave per edge
__global__ void __launch_bounds__(256) k_cheb_edge_dtg(
    const int* __restrict__ ei, const float* __restrict__ lam,
    const float* __restrict__ dinv, const float* __restrict__ src,
    float* __restrict__ dst, int t, float mul) {
  long wid = (long)blockIdx.x * 8 + (threadIdx.x >> 5);
  int lane = threadIdx.x & 31;
  if (wid >= E_EDGES) return;
  int row = ei[((long)t * 2 + 0) * E_EDGES + wid];
  int col = ei[((long)t * 2 + 1) * E_EDGES + wid];
  float wgt = mul * (-2.f / lam[t]) * dinv[(long)t * M_ROWS + row] *
              dinv[(long)t * M_ROWS + col];
  const float* sp = src + (long)col * 64;
  float* dp = dst + (long)row * 64;
  atomicAdd(&dp[lane], wgt * sp[lane]);
  atomicAdd(&dp[lane + 32], wgt * sp[lane + 32]);
}

// ---------------------------------------------------------------------------
// Fused output GEMM: acc[m,o] += sum_i Tx_i[t] @ G[t,i]   (WMMA, wave/16 rows)
// ---------------------------------------------------------------------------
__global__ void __launch_bounds__(128) k_outgemm_dtg(
    const float* __restrict__ xs_t, const float* __restrict__ Tx1,
    const float* __restrict__ Tx2, const _Float16* __restrict__ packG,
    float* __restrict__ acc, int t) {
  int wave = threadIdx.x >> 5, lane = threadIdx.x & 31;
  int tile = blockIdx.x * 4 + wave;
  if (tile * 16 >= M_ROWS) return;
  int mbase = tile * 16;
  int half = lane >> 4, colL = lane & 15;
  int rowA = mbase + colL;
  if (rowA >= M_ROWS) rowA = M_ROWS - 1;
  const float* bufs[3] = {xs_t, Tx1, Tx2};
  v8f c = zero8();
#pragma unroll
  for (int r = 0; r < 6; ++r) {
    const float* p = bufs[r >> 1] + (long)rowA * 64 + (r & 1) * 32 + half * 8;
    v4f f0 = *(const v4f*)(p);
    v4f f1 = *(const v4f*)(p + 4);
    v4f f2 = *(const v4f*)(p + 16);
    v4f f3 = *(const v4f*)(p + 20);
    v16h a;
#pragma unroll
    for (int j = 0; j < 4; ++j) {
      a[j] = (_Float16)f0[j];
      a[j + 4] = (_Float16)f1[j];
      a[j + 8] = (_Float16)f2[j];
      a[j + 12] = (_Float16)f3[j];
    }
    v16h b = *(const v16h*)(packG + ((size_t)((t * 6 + r) * 32 + lane)) * 16);
    c = wmma_f16(a, b, c);
  }
#pragma unroll
  for (int e = 0; e < 8; ++e) {
    int m = mbase + e + half * 8;
    if (m < M_ROWS && colL < 12) acc[(long)m * 12 + colL] += c[e];
  }
}

__global__ void k_final_dtg(const float* __restrict__ acc,
                            const float* __restrict__ biasO,
                            float* __restrict__ out) {
  long i = (long)blockIdx.x * blockDim.x + threadIdx.x;
  if (i >= (long)M_ROWS * 12) return;
  float x = acc[i] + biasO[i % 12];
  out[i] = 1.f / (1.f + __expf(-x));
}

// ---------------------------------------------------------------------------
// Host orchestration
// ---------------------------------------------------------------------------
extern "C" void kernel_launch(void* const* d_in, const int* in_sizes, int n_in,
                              void* d_out, int out_size, void* d_ws, size_t ws_size,
                              hipStream_t stream) {
  const float* nodef = (const float*)d_in[0];
  const int* eidx = (const int*)d_in[1];
  const float* Wih0 = (const float*)d_in[2];
  const float* Whh0 = (const float*)d_in[3];
  const float* bih0 = (const float*)d_in[4];
  const float* bhh0 = (const float*)d_in[5];
  const float* Wih1 = (const float*)d_in[6];
  const float* Whh1 = (const float*)d_in[7];
  const float* bih1 = (const float*)d_in[8];
  const float* bhh1 = (const float*)d_in[9];
  const float* lnS = (const float*)d_in[10];
  const float* lnB = (const float*)d_in[11];
  const float* chebW = (const float*)d_in[12];
  const float* chebB = (const float*)d_in[13];
  const float* convW = (const float*)d_in[14];
  const float* convB = (const float*)d_in[15];
  float* outp = (float*)d_out;

  char* ws = (char*)d_ws;
  size_t off = 0;
  auto take = [&](size_t b) -> char* {
    char* p = ws + off;
    off = (off + b + 255) & ~(size_t)255;
    return p;
  };
  float* out0 = (float*)take((size_t)M_ROWS * T_STEPS * 128 * 4);   // 184 MB
  float* out1b = (float*)take((size_t)M_ROWS * T_STEPS * 64 * 4);   //  92 MB
  float* xs = out0;  // alias: out0 dead after layer 1; xs is 92 MB <= 184 MB
  float* deg = (float*)take((size_t)T_STEPS * M_ROWS * 4);
  float* dinv = (float*)take((size_t)T_STEPS * M_ROWS * 4);
  float* vv = (float*)take((size_t)T_STEPS * M_ROWS * 4);
  float* wb = (float*)take((size_t)T_STEPS * M_ROWS * 4);
  float* sums = (float*)take(256);
  float* lam = (float*)take(256);
  float* Tx1 = (float*)take((size_t)M_ROWS * 64 * 4);
  float* Tx2 = (float*)take((size_t)M_ROWS * 64 * 4);
  float* acc = (float*)take((size_t)M_ROWS * 12 * 4);
  float* biasO = (float*)take(256);
  _Float16* pIh0 = (_Float16*)take((size_t)2 * 1 * 12 * 512 * 2);
  _Float16* pHh0 = (_Float16*)take((size_t)2 * 2 * 12 * 512 * 2);
  _Float16* pIh1 = (_Float16*)take((size_t)2 * 4 * 12 * 512 * 2);
  _Float16* pHh1 = (_Float16*)take((size_t)2 * 2 * 12 * 512 * 2);
  _Float16* pG = (_Float16*)take((size_t)72 * 512 * 2);

  const int blocksTM = (T_STEPS * M_ROWS + 255) / 256;   // 1407
  const int blocksTE = (T_STEPS * E_EDGES + 255) / 256;  // 45000
  const int blocksM64 = (M_ROWS * 64 + 255) / 256;       // 7500
  const int blocksMO = (M_ROWS * 12 + 255) / 256;        // 1407
  const int blocksGRU = (M_ROWS / 16 + 3) / 4;           // 469
  const int blocksEdge = E_EDGES / 8;                    // 120000

  // --- weight / filter pre-swizzle into WMMA fragment order ---
  k_pack_w_dtg<<<48, 256, 0, stream>>>(Wih0, pIh0, 16, 1);
  k_pack_w_dtg<<<96, 256, 0, stream>>>(Whh0, pHh0, 64, 2);
  k_pack_w_dtg<<<192, 256, 0, stream>>>(Wih1, pIh1, 128, 4);
  k_pack_w_dtg<<<96, 256, 0, stream>>>(Whh1, pHh1, 64, 2);
  k_pack_G_dtg<<<144, 256, 0, stream>>>(chebW, convW, pG);
  k_biasO_dtg<<<1, 32, 0, stream>>>(chebB, convW, convB, biasO);

  // --- BiGRU layer 0 (both dirs), layer 1 (backward only; fwd half is dead) ---
  dim3 g0(blocksGRU, 2);
  k_bigru_dtg<1><<<g0, 128, 0, stream>>>(nodef, 16L, (long)N_NODES * 16,
                                         pIh0, pHh0, bih0, bhh0, out0, 128, 64, 0);
  dim3 g1(blocksGRU, 1);
  k_bigru_dtg<4><<<g1, 128, 0, stream>>>(out0, (long)T_STEPS * 128, 128L,
                                         pIh1, pHh1, bih1, bhh1, out1b, 64, 0, 1);

  // --- LayerNorm -> time-major xs (aliases out0) ---
  k_ln_dtg<<<(M_ROWS * T_STEPS) / 4 / 32, 128, 0, stream>>>(out1b, lnS, lnB, xs);

  // --- degrees / power iteration (batched over all 12 graphs) ---
  k_zero_dtg<<<blocksTM, 256, 0, stream>>>(deg, (long)T_STEPS * M_ROWS);
  k_deg_dtg<<<blocksTE, 256, 0, stream>>>(eidx, deg);
  k_dinv_dtg<<<blocksTM, 256, 0, stream>>>(deg, dinv);
  k_fill_dtg<<<blocksTM, 256, 0, stream>>>(vv, (long)T_STEPS * M_ROWS,
                                           1.f / sqrtf((float)M_ROWS));
  dim3 gr(128, T_STEPS);
  for (int it = 0; it < 75; ++it) {
    k_zero_dtg<<<1, 32, 0, stream>>>(sums, T_STEPS);
    k_pa_dtg<<<blocksTM, 256, 0, stream>>>(deg, vv, wb);
    k_pb_dtg<<<blocksTE, 256, 0, stream>>>(eidx, vv, wb);
    k_red_dtg<<<gr, 256, 0, stream>>>(wb, wb, sums);
    k_scale_dtg<<<blocksTM, 256, 0, stream>>>(wb, sums, vv);
  }
  // lam[t] = v . L v
  k_pa_dtg<<<blocksTM, 256, 0, stream>>>(deg, vv, wb);
  k_pb_dtg<<<blocksTE, 256, 0, stream>>>(eidx, vv, wb);
  k_zero_dtg<<<1, 32, 0, stream>>>(lam, T_STEPS);
  k_red_dtg<<<gr, 256, 0, stream>>>(vv, wb, lam);

  // --- Chebyshev recursion + fused (chebW x convW) output GEMM per t ---
  k_zero_dtg<<<blocksMO, 256, 0, stream>>>(acc, (long)M_ROWS * 12);
  for (int t = 0; t < T_STEPS; ++t) {
    const float* xs_t = xs + (size_t)t * M_ROWS * 64;
    k_cheb_init1_dtg<<<blocksM64, 256, 0, stream>>>(xs_t, lam, Tx1, t);
    k_cheb_edge_dtg<<<blocksEdge, 256, 0, stream>>>(eidx, lam, dinv, xs_t, Tx1, t, 1.f);
    k_cheb_init2_dtg<<<blocksM64, 256, 0, stream>>>(xs_t, Tx1, lam, Tx2, t);
    k_cheb_edge_dtg<<<blocksEdge, 256, 0, stream>>>(eidx, lam, dinv, Tx1, Tx2, t, 2.f);
    k_outgemm_dtg<<<blocksGRU, 128, 0, stream>>>(xs_t, Tx1, Tx2, pG, acc, t);
  }
  k_final_dtg<<<blocksMO, 256, 0, stream>>>(acc, biasO, outp);
}
